// BertAttention_18494129176592
// MI455X (gfx1250) — matrix operand order repaired and checked by reading, not compiled, not run on hardware
//
#include <hip/hip_runtime.h>
#include <hip/hip_bf16.h>

// Shapes are fixed by the reference:
#define BB   2
#define SS   1024
#define HID  1024
#define HH   16
#define DD   64
#define RR   129            // 2*RELC+1
#define NEGINF (-10000.0f)

typedef __attribute__((ext_vector_type(16))) __bf16 v16bf;
typedef __attribute__((ext_vector_type(8)))  float  v8f;

union ABfrag { v16bf v; float4 f4[2]; };

__device__ __forceinline__ unsigned short f2bfu(float f) {
  union { float f; unsigned u; } in; in.f = f;
  unsigned u = in.u + 0x7fffu + ((in.u >> 16) & 1u);   // round-to-nearest-even
  return (unsigned short)(u >> 16);
}
__device__ __forceinline__ float bfu2f(unsigned short s) {
  union { unsigned u; float f; } o; o.u = ((unsigned)s) << 16;
  return o.f;
}

// ---------------- fp32 -> bf16 flat convert ----------------
__global__ void cvt_f32_bf16(const float* __restrict__ in,
                             unsigned short* __restrict__ out, int n) {
  int i = blockIdx.x * blockDim.x + threadIdx.x;
  if (i < n) out[i] = f2bfu(in[i]);
}

// ---------------- pack weight [K=1024,N=1024] into WMMA B-fragment layout ----
// layout: [(kt in 32)][(nt in 64)][(lane in 32)][(e in 16)]
// lane n = nt*16 + (lane&15); K = kt*32 + (lane>>4)*16 + e
__global__ void pack_weight(const float* __restrict__ W,
                            unsigned short* __restrict__ Wp) {
  int tid = blockIdx.x * blockDim.x + threadIdx.x;  // 32*64*32 = 65536
  int lane = tid & 31;
  int nt   = (tid >> 5) & 63;
  int kt   = tid >> 11;
  int n  = nt * 16 + (lane & 15);
  int kb = kt * 32 + ((lane >> 4) << 4);
  unsigned short* dst = Wp + (((kt * 64 + nt) * 32 + lane) << 4);
#pragma unroll
  for (int e = 0; e < 16; ++e)
    dst[e] = f2bfu(W[(kb + e) * HID + n]);
}

// ---------------- generic WMMA GEMM: C[2048,1024] = A[2048,1024] @ Bp + bias --
// B tiles are staged once per block into a 4-deep LDS ring via async
// global->LDS copies (one barrier per K-step). mode 0: fp32 row-major out.
// mode 1: bf16 head-major out [B,H,S,D].
__global__ void gemm_bf16(const unsigned short* __restrict__ A,
                          const unsigned short* __restrict__ Bp,
                          const float* __restrict__ bias,
                          void* __restrict__ out, int mode) {
  __shared__ unsigned short bstage[4][2048];   // 4 x 4KB B staging ring

  const int tid = threadIdx.x;
  const int l  = tid & 31;
  const int w  = tid >> 5;
  const int m0 = blockIdx.x * 128 + w * 16;
  const int nt0 = blockIdx.y * 4;                 // base N-tile (16 cols each)
  const int asub = ((l >> 4) << 3);               // 0 or 8
  const int arow = m0 + (l & 15);

  // Per-thread 16B slice of the 4KB B chunk for iteration kt:
  //   global element base = (kt*64 + nt0)*512 + tid*8   (chunk is contiguous)
  const unsigned short* gsrc0 = Bp + (size_t)nt0 * 512 + (size_t)tid * 8;
  unsigned ldsb[4];
#pragma unroll
  for (int q = 0; q < 4; ++q)
    ldsb[q] = (unsigned)(uintptr_t)&bstage[q][tid * 8];

  const unsigned short* arowp = A + arow * 1024;

  // prologue: stage kt=0 asynchronously
  {
    unsigned long long g = (unsigned long long)(uintptr_t)gsrc0;
    asm volatile("global_load_async_to_lds_b128 %0, %1, off"
                 :: "v"(ldsb[0]), "v"(g) : "memory");
  }

  v8f cc[4] = {};
  for (int kt = 0; kt < 31; ++kt) {
    // issue copy for kt+1 (buffer (kt+1)&3 was last read at kt-3: safe with
    // the single barrier below)
    {
      unsigned long long g =
          (unsigned long long)(uintptr_t)(gsrc0 + (size_t)(kt + 1) * 32768);
      asm volatile("global_load_async_to_lds_b128 %0, %1, off"
                   :: "v"(ldsb[(kt + 1) & 3]), "v"(g) : "memory");
    }
    // prefetch A stream two steps ahead; load this step's A fragment
    __builtin_prefetch(arowp + (kt + 2 < 32 ? (kt + 2) * 32 : 992), 0, 0);
    ABfrag a;
    a.f4[0] = *(const float4*)(arowp + kt * 32 + 2 * asub);
    a.f4[1] = *(const float4*)(arowp + kt * 32 + 2 * asub + 8);

    asm volatile("s_wait_asynccnt 0x1" ::: "memory");  // kt's copy done
    __syncthreads();

    const unsigned short* bbuf = &bstage[kt & 3][0];
#pragma unroll
    for (int j = 0; j < 4; ++j) {
      ABfrag bf;
      const unsigned short* bp = bbuf + j * 512 + l * 16;
      bf.f4[0] = *(const float4*)bp;
      bf.f4[1] = *(const float4*)(bp + 8);
      cc[j] = __builtin_amdgcn_wmma_f32_16x16x32_bf16(
          false, a.v, false, bf.v, (short)0, cc[j], false, false);
    }
  }
  // epilogue: kt = 31
  {
    ABfrag a;
    a.f4[0] = *(const float4*)(arowp + 31 * 32 + 2 * asub);
    a.f4[1] = *(const float4*)(arowp + 31 * 32 + 2 * asub + 8);
    asm volatile("s_wait_asynccnt 0x0" ::: "memory");
    __syncthreads();
    const unsigned short* bbuf = &bstage[31 & 3][0];
#pragma unroll
    for (int j = 0; j < 4; ++j) {
      ABfrag bf;
      const unsigned short* bp = bbuf + j * 512 + l * 16;
      bf.f4[0] = *(const float4*)bp;
      bf.f4[1] = *(const float4*)(bp + 8);
      cc[j] = __builtin_amdgcn_wmma_f32_16x16x32_bf16(
          false, a.v, false, bf.v, (short)0, cc[j], false, false);
    }
  }

#pragma unroll
  for (int j = 0; j < 4; ++j) {
#pragma unroll
    for (int v = 0; v < 8; ++v) {
      int row = m0 + v + asub;
      int col = (nt0 + j) * 16 + (l & 15);
      float val = cc[j][v] + bias[col];
      if (mode == 0) {
        ((float*)out)[row * 1024 + col] = val;
      } else {
        int bb = row >> 10, s = row & 1023;
        int hh = col >> 6,  d = col & 63;
        ((unsigned short*)out)[((bb * HH + hh) * SS + s) * DD + d] = f2bfu(val);
      }
    }
  }
}

// ---------------- pack k (head-major bf16) into B-fragment layout for q@k^T ---
// per (b,h): GEMM-K = D = 64 (kt in 0..1), N = S (nt in 0..63)
__global__ void pack_kT(const unsigned short* __restrict__ kh,
                        unsigned short* __restrict__ kpT) {
  int tid = blockIdx.x * blockDim.x + threadIdx.x;  // 32*2*64*32 = 131072
  int lane = tid & 31;
  int nt = (tid >> 5) & 63;
  int kt = (tid >> 11) & 1;
  int bh = tid >> 12;
  int ns = nt * 16 + (lane & 15);
  int kd = kt * 32 + ((lane >> 4) << 4);
  const float4* src = (const float4*)(kh + (bh * SS + ns) * DD + kd);
  float4* dst = (float4*)(kpT + ((((bh * 2 + kt) * 64 + nt) * 32 + lane) << 4));
  dst[0] = src[0]; dst[1] = src[1];
}

// ---------------- pack v into B-fragment layout for probs@v ------------------
// per (b,h): GEMM-K = S (kt in 0..31), N = D (nt in 0..3)
__global__ void pack_v(const unsigned short* __restrict__ vh,
                       unsigned short* __restrict__ vpk) {
  int tid = blockIdx.x * blockDim.x + threadIdx.x;  // 32*32*4*32 = 131072
  int lane = tid & 31;
  int nt = (tid >> 5) & 3;
  int kt = (tid >> 7) & 31;
  int bh = tid >> 12;
  int nd = nt * 16 + (lane & 15);
  int ks = kt * 32 + ((lane >> 4) << 4);
  unsigned short* dst = vpk + ((((bh * 32 + kt) * 4 + nt) * 32 + lane) << 4);
#pragma unroll
  for (int e = 0; e < 16; ++e)
    dst[e] = vh[(bh * SS + ks + e) * DD + nd];
}

// ---------------- q2[bhq][r] = q[bhq,:] . edge_atts[r,:] ---------------------
__global__ void q2_kernel(const unsigned short* __restrict__ qh,
                          const float* __restrict__ edge_atts,
                          float* __restrict__ q2) {
  int bhq = blockIdx.x;
  int r = threadIdx.x;
  if (r >= RR) return;
  const unsigned short* qp = qh + bhq * DD;
  const float* ep = edge_atts + r * DD;
  float acc = 0.f;
#pragma unroll 8
  for (int d = 0; d < DD; ++d) acc += bfu2f(qp[d]) * ep[d];
  q2[bhq * RR + r] = acc;
}

// ---------------- fused attention per (b,h, 16-query tile) -------------------
__global__ void attn_kernel(const unsigned short* __restrict__ qh,
                            const unsigned short* __restrict__ kpT,
                            const unsigned short* __restrict__ vpk,
                            const float* __restrict__ q2,
                            const int* __restrict__ rposi,
                            const float* __restrict__ mask_k,
                            const float* __restrict__ mask_qk,
                            const float* __restrict__ edge_values,
                            unsigned short* __restrict__ ctxo) {
  extern __shared__ char smem[];
  float*          sc   = (float*)smem;                       // 16*1024 f32 scores
  unsigned short* pb   = (unsigned short*)(sc + 16 * 1024);  // 16*1024 bf16 exp
  unsigned char*  rpcl = (unsigned char*)(pb + 16 * 1024);   // 16*1024 u8 buckets
  float* pbar   = (float*)(rpcl + 16 * 1024);                // 16*132
  float* red    = pbar + 16 * 132;                           // 256
  float* rowmax = red + 256;                                 // 16
  float* rowsum = rowmax + 16;                               // 16
  float* ctxp   = rowsum + 16;                               // 2*16*64

  const int tid = threadIdx.x;
  const int l = tid & 31, w = tid >> 5;
  const int q0 = blockIdx.x * 16;
  const int bh = blockIdx.y;
  const int b = bh >> 4, hh = bh & 15;
  const int asub = ((l >> 4) << 3);

  for (int i = tid; i < 16 * 132; i += 256) pbar[i] = 0.f;

  // Phase 1: scores tile [16 x 1024] = q @ k^T  (8 waves x 8 column tiles)
  for (int t8 = 0; t8 < 8; ++t8) {
    int nt = w * 8 + t8;
    v8f c = {};
#pragma unroll
    for (int kt = 0; kt < 2; ++kt) {
      ABfrag a, bf;
      const unsigned short* ap =
          qh + (bh * SS + q0 + (l & 15)) * DD + kt * 32 + 2 * asub;
      a.f4[0] = *(const float4*)ap;
      a.f4[1] = *(const float4*)(ap + 8);
      const unsigned short* bp =
          kpT + ((((bh * 2 + kt) * 64 + nt) * 32 + l) << 4);
      bf.f4[0] = *(const float4*)bp;
      bf.f4[1] = *(const float4*)(bp + 8);
      c = __builtin_amdgcn_wmma_f32_16x16x32_bf16(
          false, a.v, false, bf.v, (short)0, c, false, false);
    }
#pragma unroll
    for (int v = 0; v < 8; ++v)
      sc[(v + asub) * 1024 + nt * 16 + (l & 15)] = c[v];
  }
  __syncthreads();

  // Phase 2: add relative bias, scale, mask; row max
  const int r = tid >> 4, c0l = tid & 15;
  const int*   rpRow  = rposi   + (b * SS + q0 + r) * SS;
  const float* mqkRow = mask_qk + (b * SS + q0 + r) * SS;
  const float* mkRow  = mask_k  + b * SS;
  const float* q2Row  = q2 + (bh * SS + q0 + r) * RR;
  float mymax = -3.4e38f;
  for (int j = 0; j < 64; ++j) {
    int k = c0l + j * 16;
    int rp = rpRow[k];
    rp = (rp < -64 ? -64 : (rp > 64 ? 64 : rp)) + 64;
    float s = (sc[r * 1024 + k] + q2Row[rp]) * 0.125f;
    float am = mkRow[k] * mqkRow[k];
    s += (1.0f - am) * NEGINF;
    sc[r * 1024 + k] = s;
    rpcl[r * 1024 + k] = (unsigned char)rp;
    mymax = fmaxf(mymax, s);
  }
  red[tid] = mymax;
  __syncthreads();
  if (tid < 16) {
    float m = red[tid * 16];
    for (int i = 1; i < 16; ++i) m = fmaxf(m, red[tid * 16 + i]);
    rowmax[tid] = m;
  }
  __syncthreads();

  // Phase 3: unnormalized exp -> bf16 LDS, scatter into pbar, row sum
  float rm = rowmax[r], mysum = 0.f;
  for (int j = 0; j < 64; ++j) {
    int k = c0l + j * 16;
    float e = __expf(sc[r * 1024 + k] - rm);
    pb[r * 1024 + k] = f2bfu(e);
    atomicAdd(&pbar[r * 132 + rpcl[r * 1024 + k]], e);
    mysum += e;
  }
  red[tid] = mysum;
  __syncthreads();
  if (tid < 16) {
    float s = 0.f;
    for (int i = 0; i < 16; ++i) s += red[tid * 16 + i];
    rowsum[tid] = s;
  }
  __syncthreads();

  // Phase 4: ctx = exp(probs) @ v ; waves split D tiles (4) x K halves (2)
  {
    int dt = w & 3, khf = w >> 2;
    v8f c = {};
    for (int kt = khf * 16; kt < khf * 16 + 16; ++kt) {
      ABfrag a, bf;
      const unsigned short* ap = pb + (l & 15) * 1024 + kt * 32 + 2 * asub;
      a.f4[0] = *(const float4*)ap;
      a.f4[1] = *(const float4*)(ap + 8);
      const unsigned short* bp =
          vpk + ((((bh * 32 + kt) * 4 + dt) * 32 + l) << 4);
      bf.f4[0] = *(const float4*)bp;
      bf.f4[1] = *(const float4*)(bp + 8);
      c = __builtin_amdgcn_wmma_f32_16x16x32_bf16(
          false, a.v, false, bf.v, (short)0, c, false, false);
    }
#pragma unroll
    for (int v = 0; v < 8; ++v)
      ctxp[(khf * 16 + v + asub) * 64 + dt * 16 + (l & 15)] = c[v];
  }
  __syncthreads();

  // Phase 5: + pbar @ edge_values, normalize by rowsum, write bf16 [B,S,HID]
  for (int u = 0; u < 4; ++u) {
    int idx = tid + u * 256;
    int rr2 = idx >> 6, d = idx & 63;
    float acc = ctxp[rr2 * 64 + d] + ctxp[1024 + rr2 * 64 + d];
    for (int e = 0; e < RR; ++e)
      acc += pbar[rr2 * 132 + e] * edge_values[e * DD + d];
    acc /= rowsum[rr2];
    ctxo[(b * SS + q0 + rr2) * HID + hh * DD + d] = f2bfu(acc);
  }
}

// ---------------- residual + LayerNorm ---------------------------------------
__global__ void ln_kernel(const float* __restrict__ hbuf,
                          const float* __restrict__ query,
                          const float* __restrict__ g,
                          const float* __restrict__ bta,
                          float* __restrict__ out) {
  __shared__ float s1[256], s2[256];
  int row = blockIdx.x, tid = threadIdx.x;
  float x[4], sum = 0.f, sq = 0.f;
#pragma unroll
  for (int u = 0; u < 4; ++u) {
    int c = tid + u * 256;
    x[u] = hbuf[row * HID + c] + query[row * HID + c];
    sum += x[u]; sq += x[u] * x[u];
  }
  s1[tid] = sum; s2[tid] = sq;
  __syncthreads();
  for (int st = 128; st > 0; st >>= 1) {
    if (tid < st) { s1[tid] += s1[tid + st]; s2[tid] += s2[tid + st]; }
    __syncthreads();
  }
  float mean = s1[0] * (1.0f / 1024.0f);
  float var  = s2[0] * (1.0f / 1024.0f) - mean * mean;
  float rstd = rsqrtf(var + 1e-12f);
#pragma unroll
  for (int u = 0; u < 4; ++u) {
    int c = tid + u * 256;
    out[row * HID + c] = (x[u] - mean) * rstd * g[c] + bta[c];
  }
}

extern "C" void kernel_launch(void* const* d_in, const int* in_sizes, int n_in,
                              void* d_out, int out_size, void* d_ws, size_t ws_size,
                              hipStream_t stream) {
  (void)in_sizes; (void)n_in; (void)out_size; (void)ws_size;
  const float* query   = (const float*)d_in[0];
  const float* key     = (const float*)d_in[1];
  const float* value   = (const float*)d_in[2];
  const float* mask_k  = (const float*)d_in[3];
  const float* mask_qk = (const float*)d_in[4];
  const int*   rposi   = (const int*)d_in[5];
  const float* Wq = (const float*)d_in[6];
  const float* bq = (const float*)d_in[7];
  const float* Wk = (const float*)d_in[8];
  const float* bk = (const float*)d_in[9];
  const float* Wv = (const float*)d_in[10];
  const float* bv = (const float*)d_in[11];
  const float* Wd = (const float*)d_in[12];
  const float* bd = (const float*)d_in[13];
  const float* ln_g = (const float*)d_in[14];
  const float* ln_b = (const float*)d_in[15];
  const float* edge_atts   = (const float*)d_in[16];
  const float* edge_values = (const float*)d_in[17];
  float* out = (float*)d_out;

  char* ws = (char*)d_ws;
  size_t off = 0;
  auto alloc = [&](size_t bytes) -> void* {
    off = (off + 255) & ~(size_t)255;
    void* p = ws + off; off += bytes; return p;
  };
  const size_t MN = (size_t)BB * SS * HID;   // 2,097,152
  unsigned short* Xq  = (unsigned short*)alloc(MN * 2);
  unsigned short* Xk  = (unsigned short*)alloc(MN * 2);
  unsigned short* Xv  = (unsigned short*)alloc(MN * 2);
  unsigned short* Wqp = (unsigned short*)alloc((size_t)HID * HID * 2);
  unsigned short* Wkp = (unsigned short*)alloc((size_t)HID * HID * 2);
  unsigned short* Wvp = (unsigned short*)alloc((size_t)HID * HID * 2);
  unsigned short* Wdp = (unsigned short*)alloc((size_t)HID * HID * 2);
  unsigned short* qh  = (unsigned short*)alloc(MN * 2);
  unsigned short* kh  = (unsigned short*)alloc(MN * 2);
  unsigned short* vh  = (unsigned short*)alloc(MN * 2);
  unsigned short* kpT = (unsigned short*)alloc(MN * 2);
  unsigned short* vpk = (unsigned short*)alloc(MN * 2);
  float* q2   = (float*)alloc((size_t)BB * HH * SS * RR * 4);
  unsigned short* ctxo = (unsigned short*)alloc(MN * 2);
  float* hbuf = (float*)alloc(MN * 4);

  cvt_f32_bf16<<<8192, 256, 0, stream>>>(query, Xq, (int)MN);
  cvt_f32_bf16<<<8192, 256, 0, stream>>>(key,   Xk, (int)MN);
  cvt_f32_bf16<<<8192, 256, 0, stream>>>(value, Xv, (int)MN);
  pack_weight<<<256, 256, 0, stream>>>(Wq, Wqp);
  pack_weight<<<256, 256, 0, stream>>>(Wk, Wkp);
  pack_weight<<<256, 256, 0, stream>>>(Wv, Wvp);
  pack_weight<<<256, 256, 0, stream>>>(Wd, Wdp);

  gemm_bf16<<<dim3(16, 16), 256, 0, stream>>>(Xq, Wqp, bq, qh, 1);
  gemm_bf16<<<dim3(16, 16), 256, 0, stream>>>(Xk, Wkp, bk, kh, 1);
  gemm_bf16<<<dim3(16, 16), 256, 0, stream>>>(Xv, Wvp, bv, vh, 1);

  pack_kT<<<512, 256, 0, stream>>>(kh, kpT);
  pack_v <<<512, 256, 0, stream>>>(vh, vpk);
  q2_kernel<<<BB * HH * SS, 160, 0, stream>>>(qh, edge_atts, q2);

  size_t smem = 16 * 1024 * 4            // scores f32
              + 16 * 1024 * 2            // exp bf16
              + 16 * 1024                // bucket u8
              + (16 * 132 + 256 + 16 + 16 + 2 * 16 * 64) * 4;
  attn_kernel<<<dim3(SS / 16, BB * HH), 256, smem, stream>>>(
      qh, kpT, vpk, q2, rposi, mask_k, mask_qk, edge_values, ctxo);

  gemm_bf16<<<dim3(16, 16), 256, 0, stream>>>(ctxo, Wdp, bd, hbuf, 0);
  ln_kernel<<<BB * SS, 256, 0, stream>>>(hbuf, query, ln_g, ln_b, out);
}
